// MoELayer_5677946765398
// MI455X (gfx1250) — matrix operand order repaired and checked by reading, not compiled, not run on hardware
//
#include <hip/hip_runtime.h>
#include <hip/hip_bf16.h>
#include <math.h>

// ---------------------------------------------------------------------------
// MoE layer for MI455X (gfx1250, wave32, WMMA + TDM + async-LDS).
// Routed (top-2) grouped GEMMs in bf16 WMMA, fp32 accumulation.
//   275 GFLOP routed vs 1.1 PFLOP dense; ~0.6 GB traffic (~26us @ 23.3 TB/s)
//   => compute bound => keep v_wmma_f32_16x16x32_bf16 streaming; move all
//   tile staging onto TDM (tensor_load_to_lds) and async global->LDS copies,
//   double-buffered so DMA overlaps the WMMA stream.
// ---------------------------------------------------------------------------

typedef __attribute__((ext_vector_type(16))) __bf16         v16bf;
typedef __attribute__((ext_vector_type(8)))  float          v8f;
typedef __attribute__((ext_vector_type(4)))  unsigned int   u32x4;
typedef __attribute__((ext_vector_type(8)))  unsigned int   u32x8;
typedef __attribute__((ext_vector_type(4)))  float          f32x4;
typedef __attribute__((ext_vector_type(4)))  unsigned short u16x4;

#define D_TOK 1024
#define N_EXP 8
#define H2    4096
#define NTOK  8192   // 4 * 2048

// workspace layout (bytes); total required ~280 MB
#define OFF_WF   ((size_t)0)                 // [NTOK][8] f32 combine weights
#define OFF_CNT  ((size_t)262144)            // [8] int expert counts
#define OFF_BASE ((size_t)(262144 + 256))    // [9] int exclusive prefix
#define OFF_ROWS ((size_t)(262144 + 1024))   // [8][NTOK] int token ids
#define OFF_XB   (((size_t)1)  << 20)        // x  bf16 row-major        16MB
#define OFF_W1B  (((size_t)18) << 20)        // w1 bf16 swizzled blocks  64MB
#define OFF_W2B  (((size_t)84) << 20)        // w2 bf16 swizzled blocks  64MB
#define OFF_HB   (((size_t)150) << 20)       // h  bf16 row-major       128MB

static __device__ __forceinline__ unsigned short f2bf(float f) {
  unsigned int u = __builtin_bit_cast(unsigned int, f);
  u += 0x7fffu + ((u >> 16) & 1u);           // round-to-nearest-even
  return (unsigned short)(u >> 16);
}

// low 32 bits of a generic LDS pointer == LDS byte offset (ISA 10.2)
static __device__ __forceinline__ unsigned ldsAddr(const void* p) {
  return (unsigned)(size_t)p;
}

// per-lane async global->LDS 16B copy (ASYNCcnt)
static __device__ __forceinline__ void asyncCopy16(unsigned lds, unsigned long long ga) {
  asm volatile("global_load_async_to_lds_b128 %0, %1, off" :: "v"(lds), "v"(ga) : "memory");
}
static __device__ __forceinline__ void waitAsync4() {
  asm volatile("s_wait_asynccnt 0x4" ::: "memory");
}
static __device__ __forceinline__ void waitAsync0() {
  asm volatile("s_wait_asynccnt 0x0" ::: "memory");
}

// TDM: contiguous 16KB block (2048 x 8B, 1 row) global -> LDS (TENSORcnt).
// D# per ISA 8.3/8.4: group0 {count=1, lds_addr, global_addr, type=2},
// group1 {data_size=8B, tensor_dim0=tile_dim0=stride0=2048, dims1=1}.
static __device__ __forceinline__ void tdmLoad16K(unsigned ldsbyte, unsigned long long ga) {
  u32x4 g0;
  g0[0] = 1u;                                               // count=1, user mode
  g0[1] = ldsbyte;                                          // lds_addr
  g0[2] = (unsigned)ga;                                     // global_addr[31:0]
  g0[3] = ((unsigned)(ga >> 32) & 0x01ffffffu) | 0x80000000u; // addr[56:32], type=2
  u32x8 g1;
  g1[0] = (3u << 16);        // workgroup_mask=0, data_size=3 (8B)
  g1[1] = (2048u << 16);     // tensor_dim0[15:0] = 2048
  g1[2] = (1u << 16);        // tensor_dim0[31:16]=0, tensor_dim1[15:0]=1
  g1[3] = (2048u << 16);     // tensor_dim1[31:16]=0, tile_dim0 = 2048
  g1[4] = 1u;                // tile_dim1 = 1, tile_dim2 = 0
  g1[5] = 2048u;             // tensor_dim0_stride[31:0] = 2048
  g1[6] = 0u;
  g1[7] = 0u;
  asm volatile("tensor_load_to_lds %0, %1" :: "s"(g0), "s"(g1) : "memory");
}

// ---------------------------------------------------------------------------
// fp32 -> bf16 bulk convert (row-major, for x)
// ---------------------------------------------------------------------------
__global__ __launch_bounds__(256) void cvt_kernel(const float* __restrict__ src,
                                                  unsigned short* __restrict__ dst,
                                                  int n4) {
  int i = blockIdx.x * blockDim.x + threadIdx.x;
  int stride = gridDim.x * blockDim.x;
  for (; i < n4; i += stride) {
    f32x4 v = ((const f32x4*)src)[i];
    u16x4 o;
    o[0] = f2bf(v[0]); o[1] = f2bf(v[1]); o[2] = f2bf(v[2]); o[3] = f2bf(v[3]);
    ((u16x4*)dst)[i] = o;
  }
}

// ---------------------------------------------------------------------------
// fp32 -> bf16 convert + swizzle weights into fragment-ready 16KB blocks.
// Source w[e][K][N]. Block b = (e*KB + kb)*NT + nt covers K rows
// [kb*64,+64) x N cols [nt*128,+128). Inside a block (8192 bf16):
//   index = ((ks*8 + nb)*32 + khalf*16 + (col&15))*16 + ki
// so a WMMA B-fragment (lane,16 values) is one contiguous 32B run.
// One thread = one (block, ks, khalf, col) job: 16 coalesced f32 loads
// (stride N), one 32B store.
// ---------------------------------------------------------------------------
__global__ __launch_bounds__(256) void cvt_swz_kernel(const float* __restrict__ src,
                                                      unsigned short* __restrict__ dst,
                                                      int K, int N, int njobs) {
  int j = blockIdx.x * blockDim.x + threadIdx.x;
  const int stride = gridDim.x * blockDim.x;
  const int NT = N >> 7, KB = K >> 6;
  for (; j < njobs; j += stride) {
    int b   = j >> 9;
    int rr  = j & 511;
    int col = rr & 127;
    int ks  = (rr >> 7) & 1;
    int kh  = (rr >> 8) & 1;
    int nt  = b % NT;
    int ekb = b / NT;            // e*KB + kb
    int kb  = ekb % KB;
    int ee  = ekb / KB;
    int kbase = kb * 64 + ks * 32 + kh * 16;
    const float* s = src + ((size_t)ee * K + kbase) * N + nt * 128 + col;
    union { u32x4 q[2]; unsigned short h[16]; } t;
#pragma unroll
    for (int ki = 0; ki < 16; ++ki) t.h[ki] = f2bf(s[(size_t)ki * N]);
    u32x4* d = (u32x4*)(dst + (size_t)b * 8192 +
                        (size_t)(((ks * 8 + (col >> 4)) * 32 + kh * 16 + (col & 15)) * 16));
    d[0] = t.q[0];
    d[1] = t.q[1];
  }
}

// ---------------------------------------------------------------------------
// Router: one wave per token; top-2 + softmax; atomic compaction.
// ---------------------------------------------------------------------------
__global__ __launch_bounds__(256) void router_kernel(const float* __restrict__ x,
                                                     const float* __restrict__ gw,
                                                     float* __restrict__ wf,
                                                     int* __restrict__ cnt,
                                                     int* __restrict__ rows) {
  const int wave = threadIdx.x >> 5, lane = threadIdx.x & 31;
  const int token = blockIdx.x * 8 + wave;
  if (token >= NTOK) return;
  const float* xr = x + (size_t)token * D_TOK;

  float acc[N_EXP];
#pragma unroll
  for (int e = 0; e < N_EXP; ++e) acc[e] = 0.0f;
  for (int d = lane; d < D_TOK; d += 32) {
    float xv = xr[d];
#pragma unroll
    for (int e = 0; e < N_EXP; ++e) acc[e] += xv * gw[d * N_EXP + e];
  }
#pragma unroll
  for (int e = 0; e < N_EXP; ++e) {
#pragma unroll
    for (int off = 16; off > 0; off >>= 1) acc[e] += __shfl_xor(acc[e], off, 32);
  }
  if (lane == 0) {
    float m1 = acc[0]; int i1 = 0;
#pragma unroll
    for (int e = 1; e < N_EXP; ++e) if (acc[e] > m1) { m1 = acc[e]; i1 = e; }
    float m2 = -3.4e38f; int i2 = 0;
#pragma unroll
    for (int e = 0; e < N_EXP; ++e)
      if (e != i1 && acc[e] > m2) { m2 = acc[e]; i2 = e; }
    float p2 = 1.0f / (1.0f + __expf(m1 - m2));
    float p1 = 1.0f - p2;
#pragma unroll
    for (int e = 0; e < N_EXP; ++e)
      wf[(size_t)token * N_EXP + e] = (e == i1) ? p1 : ((e == i2) ? p2 : 0.0f);
    int s1 = atomicAdd(&cnt[i1], 1); rows[i1 * NTOK + s1] = token;
    int s2 = atomicAdd(&cnt[i2], 1); rows[i2 * NTOK + s2] = token;
  }
}

__global__ void prefix_kernel(const int* __restrict__ cnt, int* __restrict__ base) {
  if (threadIdx.x == 0 && blockIdx.x == 0) {
    int s = 0;
    for (int e = 0; e < N_EXP; ++e) { base[e] = s; s += cnt[e]; }
    base[N_EXP] = s;
  }
}

// ---------------------------------------------------------------------------
// GEMM1: Hg = gelu( Xg[Ne,1024] @ W1[e][1024,4096] ) -> bf16 hb.
// 128x128 tile, 8 waves x (2x4) 16x16 accum tiles, double-buffered LDS,
// A via async global->LDS (gathered rows), B via TDM (pre-swizzled block).
// ---------------------------------------------------------------------------
__global__ __launch_bounds__(256) void gemm1_kernel(const unsigned short* __restrict__ xb,
                                                    const unsigned short* __restrict__ w1s,
                                                    unsigned short* __restrict__ hb,
                                                    const int* __restrict__ cnt,
                                                    const int* __restrict__ base,
                                                    const int* __restrict__ rows) {
  const int e = blockIdx.z, mtile = blockIdx.x, ntile = blockIdx.y;
  const int count = cnt[e];
  if (mtile * 128 >= count) return;

  __shared__ __align__(16) unsigned short As[2][128][72];
  __shared__ __align__(16) unsigned short Bs[2][8192];
  __shared__ int rowIdx[128];

  const int tid = threadIdx.x;
  if (tid < 128) {
    int g = mtile * 128 + tid;
    rowIdx[tid] = (g < count) ? rows[e * NTOK + g] : rows[e * NTOK];
  }
  __syncthreads();

  const int wv = tid >> 5, lane = tid & 31;
  const int m0 = (wv & 3) * 32, n0w = (wv >> 2) * 64;
  const int ar = tid >> 3, ac0 = (tid & 7) * 8;

  auto stageA = [&](int buf, int k0) {
#pragma unroll
    for (int p = 0; p < 4; ++p) {
      int row = p * 32 + ar;
      unsigned long long ga =
          (unsigned long long)(size_t)(xb + (size_t)rowIdx[row] * D_TOK + k0 + ac0);
      asyncCopy16(ldsAddr(&As[buf][row][ac0]), ga);
    }
  };
  auto stageB = [&](int buf, int kb) {
    if (wv == 0) {
      unsigned long long ga = (unsigned long long)(size_t)(
          w1s + (((size_t)e * (D_TOK / 64) + kb) * (H2 / 128) + ntile) * 8192);
      tdmLoad16K(ldsAddr(&Bs[buf][0]), ga);
    }
  };

  stageA(0, 0);
  stageB(0, 0);

  const int NC = D_TOK / 64;
  v8f acc[2][4] = {};
  for (int kc = 0; kc < NC; ++kc) {
    const int cur = kc & 1;
    if (kc + 1 < NC) {
      stageA(cur ^ 1, (kc + 1) * 64);
      stageB(cur ^ 1, kc + 1);
      waitAsync4();
      __builtin_amdgcn_s_wait_tensorcnt(1);
    } else {
      waitAsync0();
      __builtin_amdgcn_s_wait_tensorcnt(0);
    }
    __syncthreads();
#pragma unroll
    for (int ks = 0; ks < 2; ++ks) {
      const int ksub = ks * 32;
      union { v16bf v; u32x4 q[2]; } a[2], b;
#pragma unroll
      for (int mi = 0; mi < 2; ++mi) {           // contiguous k = 16h + s A layout
        int row = m0 + mi * 16 + (lane & 15);
        int kb = ksub + ((lane >> 4) ? 16 : 0);
        a[mi].q[0] = *(const u32x4*)&As[cur][row][kb];
        a[mi].q[1] = *(const u32x4*)&As[cur][row][kb + 8];
      }
#pragma unroll
      for (int ni = 0; ni < 4; ++ni) {
        const u32x4* bp =
            (const u32x4*)&Bs[cur][(size_t)(((ks * 8 + (n0w >> 4) + ni) * 32 + lane) * 16)];
        b.q[0] = bp[0]; b.q[1] = bp[1];
#pragma unroll
        for (int mi = 0; mi < 2; ++mi) {
          acc[mi][ni] = __builtin_amdgcn_wmma_f32_16x16x32_bf16(
              false, a[mi].v, false, b.v, (short)0, acc[mi][ni], false, false);
        }
      }
    }
    __syncthreads();
  }

  const int hbase = base[e];
#pragma unroll
  for (int mi = 0; mi < 2; ++mi)
#pragma unroll
    for (int ni = 0; ni < 4; ++ni)
#pragma unroll
      for (int v = 0; v < 8; ++v) {
        int rl = m0 + mi * 16 + (lane >> 4) * 8 + v;   // C/D layout
        int g = mtile * 128 + rl;
        if (g < count) {
          float val = acc[mi][ni][v];
          val = 0.5f * val * (1.0f + erff(val * 0.70710678118f));  // exact gelu
          int col = ntile * 128 + n0w + ni * 16 + (lane & 15);
          hb[(size_t)(hbase + g) * H2 + col] = f2bf(val);
        }
      }
}

// ---------------------------------------------------------------------------
// GEMM2: Out += gate_w * ( Hg[Ne,4096] @ W2[e][4096,1024] ), f32 atomics out.
// ---------------------------------------------------------------------------
__global__ __launch_bounds__(256) void gemm2_kernel(const unsigned short* __restrict__ hb,
                                                    const unsigned short* __restrict__ w2s,
                                                    const float* __restrict__ wf,
                                                    float* __restrict__ out,
                                                    const int* __restrict__ cnt,
                                                    const int* __restrict__ base,
                                                    const int* __restrict__ rows) {
  const int e = blockIdx.z, mtile = blockIdx.x, ntile = blockIdx.y;
  const int count = cnt[e];
  if (mtile * 128 >= count) return;

  __shared__ __align__(16) unsigned short As[2][128][72];
  __shared__ __align__(16) unsigned short Bs[2][8192];
  __shared__ int   rowTok[128];
  __shared__ float rowW[128];

  const int tid = threadIdx.x;
  const int hbase = base[e];
  if (tid < 128) {
    int g = mtile * 128 + tid;
    int tok = (g < count) ? rows[e * NTOK + g] : 0;
    rowTok[tid] = tok;
    rowW[tid] = (g < count) ? wf[(size_t)tok * N_EXP + e] : 0.0f;
  }
  __syncthreads();

  const int wv = tid >> 5, lane = tid & 31;
  const int m0 = (wv & 3) * 32, n0w = (wv >> 2) * 64;
  const int ar = tid >> 3, ac0 = (tid & 7) * 8;

  auto stageA = [&](int buf, int k0) {
#pragma unroll
    for (int p = 0; p < 4; ++p) {
      int row = p * 32 + ar;
      int gl = mtile * 128 + row;
      if (gl >= count) gl = count - 1;
      unsigned long long ga =
          (unsigned long long)(size_t)(hb + (size_t)(hbase + gl) * H2 + k0 + ac0);
      asyncCopy16(ldsAddr(&As[buf][row][ac0]), ga);
    }
  };
  auto stageB = [&](int buf, int kb) {
    if (wv == 0) {
      unsigned long long ga = (unsigned long long)(size_t)(
          w2s + (((size_t)e * (H2 / 64) + kb) * (D_TOK / 128) + ntile) * 8192);
      tdmLoad16K(ldsAddr(&Bs[buf][0]), ga);
    }
  };

  stageA(0, 0);
  stageB(0, 0);

  const int NC = H2 / 64;
  v8f acc[2][4] = {};
  for (int kc = 0; kc < NC; ++kc) {
    const int cur = kc & 1;
    if (kc + 1 < NC) {
      stageA(cur ^ 1, (kc + 1) * 64);
      stageB(cur ^ 1, kc + 1);
      waitAsync4();
      __builtin_amdgcn_s_wait_tensorcnt(1);
    } else {
      waitAsync0();
      __builtin_amdgcn_s_wait_tensorcnt(0);
    }
    __syncthreads();
#pragma unroll
    for (int ks = 0; ks < 2; ++ks) {
      const int ksub = ks * 32;
      union { v16bf v; u32x4 q[2]; } a[2], b;
#pragma unroll
      for (int mi = 0; mi < 2; ++mi) {
        int row = m0 + mi * 16 + (lane & 15);
        int kb = ksub + ((lane >> 4) ? 16 : 0);
        a[mi].q[0] = *(const u32x4*)&As[cur][row][kb];
        a[mi].q[1] = *(const u32x4*)&As[cur][row][kb + 8];
      }
#pragma unroll
      for (int ni = 0; ni < 4; ++ni) {
        const u32x4* bp =
            (const u32x4*)&Bs[cur][(size_t)(((ks * 8 + (n0w >> 4) + ni) * 32 + lane) * 16)];
        b.q[0] = bp[0]; b.q[1] = bp[1];
#pragma unroll
        for (int mi = 0; mi < 2; ++mi) {
          acc[mi][ni] = __builtin_amdgcn_wmma_f32_16x16x32_bf16(
              false, a[mi].v, false, b.v, (short)0, acc[mi][ni], false, false);
        }
      }
    }
    __syncthreads();
  }

#pragma unroll
  for (int mi = 0; mi < 2; ++mi)
#pragma unroll
    for (int ni = 0; ni < 4; ++ni)
#pragma unroll
      for (int v = 0; v < 8; ++v) {
        int rl = m0 + mi * 16 + (lane >> 4) * 8 + v;
        int g = mtile * 128 + rl;
        if (g < count) {
          int tok = rowTok[rl];
          float sw = rowW[rl];
          int col = ntile * 128 + n0w + ni * 16 + (lane & 15);
          unsafeAtomicAdd(&out[(size_t)tok * D_TOK + col], sw * acc[mi][ni][v]);
        }
      }
}

// ---------------------------------------------------------------------------
extern "C" void kernel_launch(void* const* d_in, const int* in_sizes, int n_in,
                              void* d_out, int out_size, void* d_ws, size_t ws_size,
                              hipStream_t stream) {
  (void)in_sizes; (void)n_in; (void)ws_size;
  const float* x  = (const float*)d_in[0];   // [4,2048,1024]
  const float* gw = (const float*)d_in[1];   // [1024,8]
  const float* w1 = (const float*)d_in[2];   // [8,1024,4096]
  const float* w2 = (const float*)d_in[3];   // [8,4096,1024]
  float* out = (float*)d_out;

  char* ws = (char*)d_ws;
  float*          wf   = (float*)(ws + OFF_WF);
  int*            cnt  = (int*)(ws + OFF_CNT);
  int*            base = (int*)(ws + OFF_BASE);
  int*            rows = (int*)(ws + OFF_ROWS);
  unsigned short* xb   = (unsigned short*)(ws + OFF_XB);
  unsigned short* w1s  = (unsigned short*)(ws + OFF_W1B);
  unsigned short* w2s  = (unsigned short*)(ws + OFF_W2B);
  unsigned short* hb   = (unsigned short*)(ws + OFF_HB);

  hipMemsetAsync(out, 0, (size_t)out_size * sizeof(float), stream);
  hipMemsetAsync(cnt, 0, 64, stream);

  cvt_kernel<<<4096, 256, 0, stream>>>(x, xb, (NTOK * D_TOK) / 4);
  // w1: K=1024 (16 kb-blocks), N=4096 (32 n-tiles); jobs = 8*16*32*512
  cvt_swz_kernel<<<4096, 256, 0, stream>>>(w1, w1s, D_TOK, H2, N_EXP * 16 * 32 * 512);
  // w2: K=4096 (64 kb-blocks), N=1024 (8 n-tiles); jobs = 8*64*8*512
  cvt_swz_kernel<<<4096, 256, 0, stream>>>(w2, w2s, H2, D_TOK, N_EXP * 64 * 8 * 512);

  router_kernel<<<NTOK / 8, 256, 0, stream>>>(x, gw, wf, cnt, rows);
  prefix_kernel<<<1, 1, 0, stream>>>(cnt, base);

  dim3 g1(64, H2 / 128, N_EXP);
  gemm1_kernel<<<g1, 256, 0, stream>>>(xb, w1s, hb, cnt, base, rows);

  dim3 g2(64, D_TOK / 128, N_EXP);
  gemm2_kernel<<<g2, 256, 0, stream>>>(hb, w2s, wf, out, cnt, base, rows);
}
